// FunctionH_85272280695121
// MI455X (gfx1250) — compile-verified, matrix operand order checked by reading, not run on hardware
//
#include <hip/hip_runtime.h>
#include <hip/hip_bf16.h>

#define N_NODES 50000
#define N_EDGES 800000
#define D_IN    128
#define NH      8
#define F_OUT   32
#define HID     256
#define D_EMB   64
#define BINS    200
#define FEAT_LEN (HID + 2*F_OUT + D_EMB)   // 384
#define N_TILES (N_NODES/16)               // 3125 (exact)

typedef __attribute__((ext_vector_type(2)))  float  v2f;
typedef __attribute__((ext_vector_type(8)))  float  v8f;
typedef __attribute__((ext_vector_type(16))) __bf16 v16bf;

// workspace layout (float offsets)
#define OFF_VCOMB 0                         // 16*128 combined [v_src;v_trg]
#define OFF_SMS   (OFF_VCOMB + 16*D_IN)     // 8
#define OFF_BIN   (OFF_SMS + NH)            // 200*8 precomputed bin scores
#define OFF_SS    (OFF_BIN + BINS*NH)       // N*16 logits: [n][0..7]=s_src, [n][8..15]=s_trg
#define OFF_ACC_F (OFF_SS + N_NODES*16)     // int64 accumulators (8B aligned: 803656*4 % 8 == 0)
#define N_ACC     (16 + 2*NH*D_IN)          // 16 denoms + 2*8*128 u-vectors

#define ACC_SCALE     16777216.0f           // 2^24 fixed-point
#define INV_ACC_SCALE (1.0/16777216.0)

// ---------------------------------------------------------------- prep
__global__ void prep_kernel(const float* __restrict__ Wproj,
                            const float* __restrict__ a_src,
                            const float* __restrict__ a_trg,
                            const float* __restrict__ Wms,
                            const float* __restrict__ ms,
                            const float* __restrict__ dist_emb_gat,
                            const float* __restrict__ Wdist,
                            float* __restrict__ ws,
                            unsigned long long* __restrict__ acc) {
  int tid = threadIdx.x;
  // vcomb[j][d]: j<8 -> a_src[h] folded into Wproj head h ; j>=8 -> a_trg
  for (int idx = tid; idx < 16*D_IN; idx += 256) {
    int j = idx / D_IN, d = idx % D_IN, h = j & 7;
    const float* a = (j < 8) ? a_src : a_trg;
    float s = 0.f;
    for (int f = 0; f < F_OUT; ++f) s += a[h*F_OUT + f] * Wproj[(h*F_OUT + f)*D_IN + d];
    ws[OFF_VCOMB + idx] = s;
  }
  if (tid < NH) {
    float s = 0.f;
    for (int k = 0; k < HID; ++k) s += Wms[tid*HID + k] * ms[k];
    ws[OFF_SMS + tid] = s;
  }
  for (int idx = tid; idx < BINS*NH; idx += 256) {
    int b = idx / NH, h = idx % NH;
    float s = 0.f;
    for (int d = 0; d < D_EMB; ++d) s += dist_emb_gat[b*D_EMB + d] * Wdist[h*D_EMB + d];
    ws[OFF_BIN + idx] = s;
  }
  for (int i = tid; i < N_ACC; i += 256) acc[i] = 0ull;
}

// ------------------------------------------- logit GEMM via WMMA (wave32)
// S(50000x16) = X(50000x128) @ Vcomb^T(128x16); one wave per 16-row tile.
__global__ void __launch_bounds__(256)
score_wmma_kernel(const float* __restrict__ X,
                  const float* __restrict__ ws,
                  float* __restrict__ sS) {
  __shared__ float vlds[16*D_IN];          // B operand (8 KB) staged once
  for (int i = threadIdx.x; i < 16*D_IN; i += 256) vlds[i] = ws[OFF_VCOMB + i];
  __syncthreads();

  const int wave = threadIdx.x >> 5;
  const int lane = threadIdx.x & 31;
  const int tile = blockIdx.x * 8 + wave;
  if (tile >= N_TILES) return;             // uniform per wave: EXEC stays all-ones

  const int m  = lane & 15;                // A: row M ; B: col N
  const int hi = lane >> 4;                // lane group 0/1
  const float* xrow = X + (long)(tile*16 + m) * D_IN;   // A rows
  const float* vrow = vlds + m * D_IN;                  // B[K][N] = vcomb[N][K]

  v8f acc = {};
#if __has_builtin(__builtin_amdgcn_wmma_f32_16x16x4_f32)
  const int kg = hi << 1;                  // lanes 0-15: K=0,1 ; lanes 16-31: K=2,3
  for (int k = 0; k < D_IN; k += 4) {
    v2f a, b;
    a.x = xrow[k + kg]; a.y = xrow[k + kg + 1];
    b.x = vrow[k + kg]; b.y = vrow[k + kg + 1];
    acc = __builtin_amdgcn_wmma_f32_16x16x4_f32(false, a, false, b, (short)0, acc,
                                                false, false);
  }
#else
  // fallback: bf16 16x16x32 (codegen-confirmed). 16-bit A layout:
  // element e of v16bf -> K = e + 8*(e/8) + 8*(lane/16); B mirrored with N=lane%16.
  for (int k = 0; k < D_IN; k += 32) {
    v16bf a, b;
    #pragma unroll
    for (int e = 0; e < 16; ++e) {
      int kk = k + e + ((e >> 3) << 3) + (hi << 3);
      a[e] = (__bf16)xrow[kk];
      b[e] = (__bf16)vrow[kk];
    }
    acc = __builtin_amdgcn_wmma_f32_16x16x32_bf16(false, a, false, b, (short)0, acc,
                                                  false, false);
  }
#endif
  // C/D layout: VGPR j -> M = j + 8*(lane/16), N = lane%16
  const int rbase = tile*16 + (hi << 3);
  #pragma unroll
  for (int j = 0; j < 8; ++j)
    sS[(long)(rbase + j)*16 + m] = acc[j];
}

// --------------------------------- edge scan: accumulate denom & u (fixed-point)
__global__ void edge_kernel(const int* __restrict__ esrc,
                            const int* __restrict__ etrg,
                            const int* __restrict__ ebin,
                            const float* __restrict__ X,
                            const int* __restrict__ pli,
                            const int* __restrict__ pld,
                            const float* __restrict__ ws,
                            unsigned long long* __restrict__ acc) {
  const int li = pli[0], ld = pld[0];
  const float* sS    = ws + OFF_SS;
  const float* sms   = ws + OFF_SMS;
  const float* binsc = ws + OFF_BIN;
  const int stride = gridDim.x * blockDim.x;
  for (int e = blockIdx.x * blockDim.x + threadIdx.x; e < N_EDGES; e += stride) {
    const int t = etrg[e];
    const bool m0 = (t == li), m1 = (t == ld);
    if (!(m0 | m1)) continue;
    const int src = esrc[e];
    const int bin = ebin[e];
    float ex[NH];
    #pragma unroll
    for (int h = 0; h < NH; ++h) {
      float sc = sS[(long)src*16 + h] + sS[(long)t*16 + 8 + h]
               + sms[h] + binsc[bin*NH + h];
      sc = (sc > 0.f) ? sc : 0.2f * sc;              // leaky_relu(0.2)
      ex[h] = expf(sc);                              // softmax shift dropped (invariant)
    }
    #pragma unroll
    for (int slot = 0; slot < 2; ++slot) {
      if (!(slot ? m1 : m0)) continue;
      #pragma unroll
      for (int h = 0; h < NH; ++h)
        atomicAdd(&acc[slot*NH + h],
                  (unsigned long long)(long long)llrintf(ex[h] * ACC_SCALE));
      for (int d = 0; d < D_IN; ++d) {
        const float xv = X[(long)src*D_IN + d];
        #pragma unroll
        for (int h = 0; h < NH; ++h)
          atomicAdd(&acc[16 + (slot*NH + h)*D_IN + d],
                    (unsigned long long)(long long)llrintf(ex[h] * xv * ACC_SCALE));
      }
    }
  }
}

// ---------------------------------------------------------------- finalize
__global__ void __launch_bounds__(256)
finalize_kernel(const float* __restrict__ X,
                const float* __restrict__ ms,
                const int* __restrict__ pli, const int* __restrict__ pld,
                const int* __restrict__ pdist,
                const float* __restrict__ Wproj, const float* __restrict__ Wskip,
                const float* __restrict__ gat_bias,
                const float* __restrict__ dist_emb_h,
                const float* __restrict__ W1, const float* __restrict__ b1,
                const float* __restrict__ W2, const float* __restrict__ b2,
                const unsigned long long* __restrict__ acc,
                float* __restrict__ out) {
  __shared__ float u_sh[2*NH*D_IN];     // 2048
  __shared__ float den_sh[16];
  __shared__ float outm[2*NH*F_OUT];    // 512
  __shared__ float emb[2*F_OUT];        // 64
  __shared__ float feat[FEAT_LEN];      // 384
  __shared__ float h1[HID];
  __shared__ float red[256];
  const int tid = threadIdx.x;

  for (int i = tid; i < 2*NH*D_IN; i += 256)
    u_sh[i] = (float)((double)(long long)acc[16 + i] * INV_ACC_SCALE);
  if (tid < 16)
    den_sh[tid] = (float)((double)(long long)acc[tid] * INV_ACC_SCALE);
  __syncthreads();

  const int li = pli[0], ld = pld[0], dist = pdist[0];

  // out[t,h,f] = (Wproj_h @ u[t,h]) / denom  +  (Wskip_h @ x[node_t])
  for (int idx = tid; idx < 2*NH*F_OUT; idx += 256) {
    const int slot = idx / (NH*F_OUT);
    const int hf = idx % (NH*F_OUT), h = hf / F_OUT, f = hf % F_OUT;
    const int node = slot ? ld : li;
    const float* wp = Wproj + (h*F_OUT + f)*D_IN;
    const float* wk = Wskip + (h*F_OUT + f)*D_IN;
    const float* uu = u_sh + (slot*NH + h)*D_IN;
    const float* xn = X + (long)node*D_IN;
    float v = 0.f, sk = 0.f;
    for (int d = 0; d < D_IN; ++d) { v += wp[d]*uu[d]; sk += wk[d]*xn[d]; }
    outm[idx] = v / (den_sh[slot*NH + h] + 1e-16f) + sk;
  }
  __syncthreads();

  if (tid < 2*F_OUT) {                          // head mean + bias + ELU
    const int slot = tid / F_OUT, f = tid % F_OUT;
    float s = 0.f;
    for (int h = 0; h < NH; ++h) s += outm[slot*NH*F_OUT + h*F_OUT + f];
    s = s * (1.f/NH) + gat_bias[f];
    emb[tid] = (s > 0.f) ? s : (expf(s) - 1.f);
  }
  for (int k = tid; k < HID; k += 256) feat[k] = ms[k];
  if (tid < 2*F_OUT) feat[HID + tid] = emb[tid];          // same-thread as emb write
  if (tid < D_EMB)   feat[HID + 2*F_OUT + tid] = dist_emb_h[dist*D_EMB + tid];
  __syncthreads();

  {                                             // h1 = relu(W1 @ feat + b1)
    float s = b1[tid];
    const float* w = W1 + (long)tid * FEAT_LEN;
    for (int k = 0; k < FEAT_LEN; ++k) s += w[k]*feat[k];
    h1[tid] = (s > 0.f) ? s : 0.f;
  }
  __syncthreads();

  red[tid] = W2[tid] * h1[tid];
  __syncthreads();
  for (int s = 128; s > 0; s >>= 1) {
    if (tid < s) red[tid] += red[tid + s];
    __syncthreads();
  }
  if (tid == 0) out[0] = red[0] + b2[0];
}

// ---------------------------------------------------------------- launch
extern "C" void kernel_launch(void* const* d_in, const int* in_sizes, int n_in,
                              void* d_out, int out_size, void* d_ws, size_t ws_size,
                              hipStream_t stream) {
  const float* X      = (const float*)d_in[0];
  const int*   esrc   = (const int*)  d_in[1];
  const int*   etrg   = (const int*)  d_in[2];
  const int*   ebin   = (const int*)  d_in[3];
  const float* ms     = (const float*)d_in[4];
  const int*   pli    = (const int*)  d_in[5];
  const int*   pld    = (const int*)  d_in[6];
  const int*   pdist  = (const int*)  d_in[7];
  const float* Wproj  = (const float*)d_in[8];
  const float* a_src  = (const float*)d_in[9];
  const float* a_trg  = (const float*)d_in[10];
  const float* Wms    = (const float*)d_in[11];
  const float* demb_g = (const float*)d_in[12];
  const float* Wdist  = (const float*)d_in[13];
  const float* Wskip  = (const float*)d_in[14];
  const float* gbias  = (const float*)d_in[15];
  const float* demb_h = (const float*)d_in[16];
  const float* W1     = (const float*)d_in[17];
  const float* b1     = (const float*)d_in[18];
  const float* W2     = (const float*)d_in[19];
  const float* b2     = (const float*)d_in[20];

  float* ws = (float*)d_ws;
  unsigned long long* acc = (unsigned long long*)(ws + OFF_ACC_F);
  float* sS = ws + OFF_SS;

  prep_kernel<<<1, 256, 0, stream>>>(Wproj, a_src, a_trg, Wms, ms, demb_g, Wdist, ws, acc);
  score_wmma_kernel<<<(N_TILES + 7)/8, 256, 0, stream>>>(X, ws, sS);
  edge_kernel<<<512, 256, 0, stream>>>(esrc, etrg, ebin, X, pli, pld, ws, acc);
  finalize_kernel<<<1, 256, 0, stream>>>(X, ms, pli, pld, pdist, Wproj, Wskip, gbias,
                                         demb_h, W1, b1, W2, b2, acc, (float*)d_out);
}
